// Model_23295902614331
// MI455X (gfx1250) — matrix operand order repaired and checked by reading, not compiled, not run on hardware
//
#include <hip/hip_runtime.h>
#include <hip/hip_bf16.h>
#include <math.h>

// ---------------------------------------------------------------------------
// PointMLP-style classifier forward for MI455X (gfx1250), wave32 + WMMA bf16.
// All heavy 1x1 convs run through v_wmma_f32_16x16x32_bf16 (f32 accumulate).
// Activations: point-major [L, Cpad] bf16, Cpad % 32 == 0 -> every WMMA
// fragment load is a contiguous global_load_b128.
// ---------------------------------------------------------------------------

typedef __bf16 bf16;
typedef bf16  bf16x16 __attribute__((ext_vector_type(16)));
typedef float f32x8   __attribute__((ext_vector_type(8)));

#define EPS_INV 0.9999950000374996f /* 1/sqrt(1+1e-5) for eval-mode BN */

union FragU { bf16x16 v; uint4 u[2]; };
union Out8U { bf16 h[8]; uint4 u; };

// ---------------------------------------------------------------------------
// Fused GEMM: Y[p, m] = relu?( (sum_k W[m,k] * X[p,k]) + b[m]) * (g[m]*EPS_INV)
//                        + bb[m] (+ res[p,m]) )
// W: [M, Kp] bf16 row-major (zero padded).  X: [L, Kp] bf16 point-major.
// Block = 8 waves; wave w owns output tile rows m0=blockIdx.y*128+w*16,
// cols n0=blockIdx.x*16.  K-loop in steps of 32 via WMMA f32_16x16x32_bf16.
//
// A (16x32 bf16) lane layout: lanes 0-15 row M=lane hold K[k..k+7],K[k+16..23];
// lanes 16-31 same rows hold K[k+8..15],K[k+24..31]  -> two b128 loads/lane.
// B (32x16 bf16): lanes 0-15 col N=lane hold K[k..k+7],K[k+16..23]; lanes 16-31
// col N=lane-16 hold K[k+8..15],K[k+24..31] -> two b128 loads/lane.
// D (16x16 f32): lanes 0-15 col N=lane, rows m0..m0+7; lanes 16-31 col
// N=lane-16, rows m0+8..m0+15 -> one contiguous 8-channel store per lane.
// ---------------------------------------------------------------------------
__global__ __launch_bounds__(256) void k_gemm(
    const bf16* __restrict__ W, const bf16* __restrict__ X,
    const float* __restrict__ bias, const float* __restrict__ gam,
    const float* __restrict__ bet, const bf16* __restrict__ res,
    bf16* __restrict__ Y, int M, int Kp, int relu)
{
  const int wave = threadIdx.x >> 5;
  const int lane = threadIdx.x & 31;
  const int half = lane >> 4;
  const int lrow = lane & 15;
  const int m0 = blockIdx.y * 128 + wave * 16;
  if (m0 >= M) return;                       // wave-uniform guard
  const long p = (long)blockIdx.x * 16 + lrow;

  const bf16* wrow = W + (long)(m0 + lrow) * Kp + half * 8;
  const bf16* xrow = X + p * (long)Kp + half * 8;

  f32x8 acc = {};
  for (int k = 0; k < Kp; k += 32) {
    FragU a, b;
    a.u[0] = *(const uint4*)(wrow + k);
    a.u[1] = *(const uint4*)(wrow + k + 16);
    b.u[0] = *(const uint4*)(xrow + k);
    b.u[1] = *(const uint4*)(xrow + k + 16);
    acc = __builtin_amdgcn_wmma_f32_16x16x32_bf16(
        false, a.v, false, b.v, (short)0, acc, false, false);
  }

  const int ch0 = m0 + half * 8;
  const long ybase = p * (long)M + ch0;
  Out8U o;
#pragma unroll
  for (int r = 0; r < 8; ++r) {
    const int ch = ch0 + r;
    float v = acc[r] + bias[ch];
    v = v * (gam[ch] * EPS_INV) + bet[ch];
    if (res) v += (float)res[ybase + r];
    if (relu && v < 0.f) v = 0.f;
    o.h[r] = (bf16)v;
  }
  *(uint4*)(Y + ybase) = o.u;
}

// fp32 [M,K] weight -> bf16 [M,Kp] zero-padded
__global__ void k_convert_w(const float* __restrict__ src, bf16* __restrict__ dst,
                            int M, int K, int Kp)
{
  long id = (long)blockIdx.x * blockDim.x + threadIdx.x;
  if (id >= (long)M * Kp) return;
  int c = (int)(id % Kp);
  long m = id / Kp;
  dst[id] = (c < K) ? (bf16)src[m * K + c] : (bf16)0.f;
}

// x [B,3,N] -> xyz [B,N,3] f32 and embed input [B*N, 32] bf16 (3 ch + zero pad)
__global__ void k_build_embed(const float* __restrict__ x, bf16* __restrict__ Xe,
                              float* __restrict__ xyz, int B, int N)
{
  long id = (long)blockIdx.x * blockDim.x + threadIdx.x;
  if (id >= (long)B * N) return;
  int n = (int)(id % N);
  int b = (int)(id / N);
  float px = x[((long)b * 3 + 0) * N + n];
  float py = x[((long)b * 3 + 1) * N + n];
  float pz = x[((long)b * 3 + 2) * N + n];
  xyz[id * 3 + 0] = px; xyz[id * 3 + 1] = py; xyz[id * 3 + 2] = pz;
  bf16* r = Xe + id * 32;
  r[0] = (bf16)px; r[1] = (bf16)py; r[2] = (bf16)pz;
  for (int c = 3; c < 32; ++c) r[c] = (bf16)0.f;
}

// Farthest point sampling; one block per batch, serial over s iterations,
// LDS-resident distance array + argmax reduction (ties -> lowest index).
__global__ __launch_bounds__(256) void k_fps(const float* __restrict__ xyz,
                                             int* __restrict__ idx, int Ncur, int s)
{
  __shared__ float dist[1024];
  __shared__ float rval[256];
  __shared__ int   ridx[256];
  __shared__ int   sfar;
  const int b = blockIdx.x, t = threadIdx.x;
  const float* P = xyz + (long)b * Ncur * 3;
  for (int n = t; n < Ncur; n += 256) dist[n] = 1e10f;
  if (t == 0) sfar = 0;
  __syncthreads();
  for (int i = 0; i < s; ++i) {
    const int far = sfar;
    if (t == 0) idx[b * s + i] = far;
    const float cx = P[far * 3], cy = P[far * 3 + 1], cz = P[far * 3 + 2];
    float bv = -3.0e38f; int bi = 0x7fffffff;
    for (int n = t; n < Ncur; n += 256) {
      float dx = P[n * 3] - cx, dy = P[n * 3 + 1] - cy, dz = P[n * 3 + 2] - cz;
      float d = dx * dx + dy * dy + dz * dz;
      float m = fminf(dist[n], d);
      dist[n] = m;
      if (m > bv) { bv = m; bi = n; }
    }
    rval[t] = bv; ridx[t] = bi;
    __syncthreads();
    for (int off = 128; off > 0; off >>= 1) {
      if (t < off) {
        if (rval[t + off] > rval[t] ||
            (rval[t + off] == rval[t] && ridx[t + off] < ridx[t])) {
          rval[t] = rval[t + off]; ridx[t] = ridx[t + off];
        }
      }
      __syncthreads();
    }
    if (t == 0) sfar = ridx[0];
    __syncthreads();
  }
}

__global__ void k_gather_xyz(const float* __restrict__ xyz, const int* __restrict__ idx,
                             float* __restrict__ xyz_s, int Ncur, int s, int total)
{
  int id = blockIdx.x * blockDim.x + threadIdx.x;
  if (id >= total) return;
  int b = id / s;
  int n = idx[id];
  const float* p = xyz + ((long)b * Ncur + n) * 3;
  xyz_s[id * 3 + 0] = p[0]; xyz_s[id * 3 + 1] = p[1]; xyz_s[id * 3 + 2] = p[2];
}

// 32-NN per query (replace-current-worst selection; set is order-invariant
// downstream: mean over K, symmetric conv + max over K).
__global__ __launch_bounds__(64) void k_knn(const float* __restrict__ xyz,
                                            const float* __restrict__ q,
                                            int* __restrict__ gidx,
                                            int Ncur, int s, int total)
{
  int id = blockIdx.x * blockDim.x + threadIdx.x;
  if (id >= total) return;
  int b = id / s;
  float qx = q[id * 3], qy = q[id * 3 + 1], qz = q[id * 3 + 2];
  float bd[32]; int bix[32];
#pragma unroll
  for (int j = 0; j < 32; ++j) { bd[j] = 3.0e38f; bix[j] = 0; }
  float worst = 3.0e38f; int wp = 0;
  const float* P = xyz + (long)b * Ncur * 3;
  for (int n = 0; n < Ncur; ++n) {
    float dx = P[n * 3] - qx, dy = P[n * 3 + 1] - qy, dz = P[n * 3 + 2] - qz;
    float d = dx * dx + dy * dy + dz * dz;
    if (d < worst) {
      bd[wp] = d; bix[wp] = n;
      worst = bd[0]; wp = 0;
#pragma unroll
      for (int j = 1; j < 32; ++j) if (bd[j] > worst) { worst = bd[j]; wp = j; }
    }
  }
  int* o = gidx + (long)id * 32;
#pragma unroll
  for (int j = 0; j < 32; ++j) o[j] = bix[j];
}

__global__ void k_zero_f32(float* p, int n)
{
  int i = blockIdx.x * blockDim.x + threadIdx.x;
  if (i < n) p[i] = 0.f;
}

// Per-(b,s,c) mean over K and per-batch sum of squared diffs (atomic).
__global__ void k_group_stats(const bf16* __restrict__ f, int ldf,
                              const float* __restrict__ xyz,
                              const int* __restrict__ gidx,
                              float* __restrict__ mean, float* __restrict__ ssq,
                              int Ncur, int s, int d, long total)
{
  long id = (long)blockIdx.x * blockDim.x + threadIdx.x;
  if (id >= total) return;
  const int dc = d + 3;
  const int c = (int)(id % dc);
  const long g = id / dc;
  const int b = (int)(g / s);
  const int* gi = gidx + g * 32;
  float sum = 0.f;
  for (int k = 0; k < 32; ++k) {
    int n = gi[k];
    float v = (c < d) ? (float)f[((long)b * Ncur + n) * ldf + c]
                      : xyz[((long)b * Ncur + n) * 3 + (c - d)];
    sum += v;
  }
  float mn = sum * (1.f / 32.f);
  mean[id] = mn;
  float ss = 0.f;
  for (int k = 0; k < 32; ++k) {
    int n = gi[k];
    float v = (c < d) ? (float)f[((long)b * Ncur + n) * ldf + c]
                      : xyz[((long)b * Ncur + n) * 3 + (c - d)];
    float t = v - mn;
    ss += t * t;
  }
  atomicAdd(&ssq[b], ss);
}

// Build normalized group rows [B*s*K, Kp] bf16:
//   [0, d+3)    : alpha*((raw-mean)/(std+1e-5)) + beta
//   [d+3, 2d+3) : anchor features f_s (broadcast over k)
//   [2d+3, Kp)  : zero pad
__global__ void k_group_build(const bf16* __restrict__ f, int ldf,
                              const float* __restrict__ xyz,
                              const int* __restrict__ gidx,
                              const int* __restrict__ fpsidx,
                              const float* __restrict__ mean,
                              const float* __restrict__ ssq,
                              const float* __restrict__ alpha,
                              const float* __restrict__ beta,
                              bf16* __restrict__ G, int Kp,
                              int Ncur, int s, int d, float inv_cnt, long total)
{
  long row = (long)blockIdx.x * blockDim.x + threadIdx.x;
  if (row >= total) return;
  const int k = (int)(row & 31);
  const long g = row >> 5;
  const int si = (int)(g % s);
  const int b  = (int)(g / s);
  const int n  = gidx[g * 32 + k];
  const float inv = 1.f / (sqrtf(ssq[b] * inv_cnt) + 1e-5f);
  const float* mn = mean + g * (d + 3);
  const long src = ((long)b * Ncur + n) * ldf;
  const long anc = ((long)b * Ncur + fpsidx[(long)b * s + si]) * ldf;
  bf16* o = G + row * (long)Kp;
  const int dc = d + 3;
  for (int c = 0; c < dc; ++c) {
    float v = (c < d) ? (float)f[src + c] : xyz[((long)b * Ncur + n) * 3 + (c - d)];
    o[c] = (bf16)(alpha[c] * ((v - mn[c]) * inv) + beta[c]);
  }
  for (int c = 0; c < d; ++c) o[dc + c] = f[anc + c];
  for (int c = 2 * d + 3; c < Kp; ++c) o[c] = (bf16)0.f;
}

// Max over K=32 consecutive rows: [rows_s*32, C] -> [rows_s, C]
__global__ void k_maxpool_k(const bf16* __restrict__ Y, bf16* __restrict__ out,
                            int C, long total)
{
  long id = (long)blockIdx.x * blockDim.x + threadIdx.x;
  if (id >= total) return;
  const int c = (int)(id % C);
  const long rs = id / C;
  const bf16* src = Y + (rs * 32) * (long)C + c;
  float m = -3.0e38f;
  for (int k = 0; k < 32; ++k) {
    float v = (float)src[(long)k * C];
    if (v > m) m = v;
  }
  out[id] = (bf16)m;
}

// Global max over s rows per batch: [B*s, C] -> [B, C] bf16
__global__ void k_global_pool(const bf16* __restrict__ f, bf16* __restrict__ gb,
                              int s, int C, int total)
{
  int id = blockIdx.x * blockDim.x + threadIdx.x;
  if (id >= total) return;
  const int c = id % C;
  const int b = id / C;
  float m = -3.0e38f;
  for (int r = 0; r < s; ++r) {
    float v = (float)f[((long)b * s + r) * C + c];
    if (v > m) m = v;
  }
  gb[id] = (bf16)m;
}

// Final small linear 256 -> 40 in f32 directly to d_out
__global__ void k_final(const bf16* __restrict__ act, const float* __restrict__ w3,
                        const float* __restrict__ b3, float* __restrict__ out,
                        int Bn, int Cin, int Cout)
{
  int id = blockIdx.x * blockDim.x + threadIdx.x;
  if (id >= Bn * Cout) return;
  const int j = id % Cout;
  const int b = id / Cout;
  float sum = b3[j];
  for (int c = 0; c < Cin; ++c) sum += (float)act[(long)b * Cin + c] * w3[(long)j * Cin + c];
  out[id] = sum;
}

// ---------------------------------------------------------------------------
struct Conv { const float *b, *bb, *g, *w; };

extern "C" void kernel_launch(void* const* d_in, const int* in_sizes, int n_in,
                              void* d_out, int out_size, void* d_ws, size_t ws_size,
                              hipStream_t stream)
{
  (void)in_sizes; (void)n_in; (void)out_size; (void)ws_size;
  const int B = 32, N0 = 1024, KNB = 32;
  const int ch[5] = {64, 128, 256, 512, 1024};

  // ---- input leaves (jax pytree order: dict keys sorted, lists in order) ---
  auto F = [&](int i) { return (const float*)d_in[i]; };
  const float* cls_b3 = F(0);
  Conv l1 = {F(1), F(2), F(3), F(4)};
  Conv l2 = {F(5), F(6), F(7), F(8)};
  const float* cls_w3 = F(9);
  Conv embed = {F(10), F(11), F(12), F(13)};
  Conv tr[4], pre[4][4], pos[4][4];
  const float *alpha[4], *beta[4];
  for (int i = 0; i < 4; ++i) {
    int base = 14 + 38 * i;   // alpha,beta | pos(2 res x 2 conv) | pre(..) | transfer
    alpha[i] = F(base + 0);
    beta[i]  = F(base + 1);
    for (int r = 0; r < 4; ++r)
      pos[i][r] = Conv{F(base + 2 + 4 * r), F(base + 3 + 4 * r),
                       F(base + 4 + 4 * r), F(base + 5 + 4 * r)};
    for (int r = 0; r < 4; ++r)
      pre[i][r] = Conv{F(base + 18 + 4 * r), F(base + 19 + 4 * r),
                       F(base + 20 + 4 * r), F(base + 21 + 4 * r)};
    tr[i] = Conv{F(base + 34), F(base + 35), F(base + 36), F(base + 37)};
  }
  const float* x = F(14 + 38 * 4);  // leaf 166: input points [B,3,N]

  // ---- workspace carve (bump allocator, 256B aligned; ~480 MB total) ------
  char* wsp = (char*)d_ws;
  size_t off = 0;
  auto alloc = [&](size_t bytes) -> char* {
    char* r = wsp + off;
    off = (off + bytes + 255) & ~(size_t)255;
    return r;
  };
  auto conv_w = [&](const float* src, int M, int K, int Kp) -> bf16* {
    bf16* dst = (bf16*)alloc((size_t)M * Kp * sizeof(bf16));
    long tot = (long)M * Kp;
    k_convert_w<<<dim3((unsigned)((tot + 255) / 256)), dim3(256), 0, stream>>>(src, dst, M, K, Kp);
    return dst;
  };
  auto gemm = [&](const bf16* W, const bf16* X, const Conv& c, const bf16* res,
                  bf16* Y, int M, int Kp, long L, int relu) {
    dim3 grid((unsigned)(L / 16), (unsigned)((M + 127) / 128));
    k_gemm<<<grid, dim3(256), 0, stream>>>(W, X, c.b, c.g, c.bb, res, Y, M, Kp, relu);
  };

  // weights -> bf16, zero-padded K
  int inch[4], Kpi[4];
  bf16 *trW[4], *preW[4][4], *posW[4][4];
  bf16* embW = conv_w(embed.w, 64, 3, 32);
  for (int i = 0; i < 4; ++i) {
    int d = ch[i], out = ch[i + 1];
    inch[i] = 2 * d + 3;
    Kpi[i] = ((inch[i] + 31) / 32) * 32;
    trW[i] = conv_w(tr[i].w, out, inch[i], Kpi[i]);
    for (int r = 0; r < 4; ++r) preW[i][r] = conv_w(pre[i][r].w, out, out, out);
    for (int r = 0; r < 4; ++r) posW[i][r] = conv_w(pos[i][r].w, out, out, out);
  }
  bf16* l1W = conv_w(l1.w, 512, 1024, 1024);
  bf16* l2W = conv_w(l2.w, 256, 512, 512);

  // activation arenas
  bf16*  Xe    = (bf16*)alloc((size_t)B * N0 * 32 * sizeof(bf16));
  bf16*  fA    = (bf16*)alloc((size_t)2097152 * sizeof(bf16));  // max rows*ld for f
  bf16*  fT    = (bf16*)alloc((size_t)2097152 * sizeof(bf16));
  bf16*  G     = (bf16*)alloc((size_t)524288 * 160 * sizeof(bf16));
  bf16*  Y1    = (bf16*)alloc((size_t)67108864 * sizeof(bf16));
  bf16*  Y2    = (bf16*)alloc((size_t)67108864 * sizeof(bf16));
  float* xyzA  = (float*)alloc((size_t)B * N0 * 3 * sizeof(float));
  float* xyzB  = (float*)alloc((size_t)B * N0 * 3 * sizeof(float));
  int*   fpsIdx = (int*)alloc((size_t)B * 512 * sizeof(int));
  int*   gidxB  = (int*)alloc((size_t)B * 512 * 32 * sizeof(int));
  float* meanB  = (float*)alloc((size_t)1097728 * sizeof(float));
  float* ssq    = (float*)alloc((size_t)B * sizeof(float));
  bf16*  gb     = (bf16*)alloc((size_t)B * 1024 * sizeof(bf16));
  bf16*  cb1    = (bf16*)alloc((size_t)B * 512 * sizeof(bf16));
  bf16*  cb2    = (bf16*)alloc((size_t)B * 256 * sizeof(bf16));

  // ---- pipeline -----------------------------------------------------------
  {
    long tot = (long)B * N0;
    k_build_embed<<<dim3((unsigned)((tot + 255) / 256)), dim3(256), 0, stream>>>(x, Xe, xyzA, B, N0);
  }
  gemm(embW, Xe, embed, nullptr, fA, 64, 32, (long)B * N0, 1);

  float* xyz_cur = xyzA;
  float* xyz_nxt = xyzB;
  bf16* f = fA;
  for (int i = 0; i < 4; ++i) {
    const int Ncur = N0 >> i, s = Ncur >> 1, d = ch[i], out = ch[i + 1];
    const long rows = (long)B * s * KNB;

    k_fps<<<dim3(B), dim3(256), 0, stream>>>(xyz_cur, fpsIdx, Ncur, s);
    {
      int tot = B * s;
      k_gather_xyz<<<dim3((tot + 255) / 256), dim3(256), 0, stream>>>(xyz_cur, fpsIdx, xyz_nxt, Ncur, s, tot);
      k_knn<<<dim3((tot + 63) / 64), dim3(64), 0, stream>>>(xyz_cur, xyz_nxt, gidxB, Ncur, s, tot);
    }
    k_zero_f32<<<dim3(1), dim3(32), 0, stream>>>(ssq, B);
    {
      long tot = (long)B * s * (d + 3);
      k_group_stats<<<dim3((unsigned)((tot + 255) / 256)), dim3(256), 0, stream>>>(
          f, d, xyz_cur, gidxB, meanB, ssq, Ncur, s, d, tot);
    }
    const float inv_cnt = 1.f / ((float)s * KNB * (d + 3) - 1.f);  // ddof=1
    k_group_build<<<dim3((unsigned)((rows + 255) / 256)), dim3(256), 0, stream>>>(
        f, d, xyz_cur, gidxB, fpsIdx, meanB, ssq, alpha[i], beta[i],
        G, Kpi[i], Ncur, s, d, inv_cnt, rows);

    gemm(trW[i], G, tr[i], nullptr, Y1, out, Kpi[i], rows, 1);
    for (int r = 0; r < 2; ++r) {                                  // pre residuals
      gemm(preW[i][2 * r], Y1, pre[i][2 * r], nullptr, Y2, out, out, rows, 1);
      gemm(preW[i][2 * r + 1], Y2, pre[i][2 * r + 1], Y1, Y1, out, out, rows, 1);
    }
    {
      long tot = (long)B * s * out;
      k_maxpool_k<<<dim3((unsigned)((tot + 255) / 256)), dim3(256), 0, stream>>>(Y1, f, out, tot);
    }
    for (int r = 0; r < 2; ++r) {                                  // pos residuals
      gemm(posW[i][2 * r], f, pos[i][2 * r], nullptr, fT, out, out, (long)B * s, 1);
      gemm(posW[i][2 * r + 1], fT, pos[i][2 * r + 1], f, f, out, out, (long)B * s, 1);
    }
    float* t = xyz_cur; xyz_cur = xyz_nxt; xyz_nxt = t;
  }

  {
    int tot = B * 1024;
    k_global_pool<<<dim3((tot + 255) / 256), dim3(256), 0, stream>>>(f, gb, 64, 1024, tot);
  }
  gemm(l1W, gb, l1, nullptr, cb1, 512, 1024, B, 1);
  gemm(l2W, cb1, l2, nullptr, cb2, 256, 512, B, 1);
  {
    int tot = B * 40;
    k_final<<<dim3((tot + 255) / 256), dim3(256), 0, stream>>>(cb2, cls_w3, cls_b3, (float*)d_out, B, 256, 40);
  }
}
